// SelfAttentionBlock_25537875542491
// MI455X (gfx1250) — matrix operand order repaired and checked by reading, not compiled, not run on hardware
//
#include <hip/hip_runtime.h>

typedef __attribute__((ext_vector_type(16))) _Float16 v16h;
typedef __attribute__((ext_vector_type(8)))  _Float16 v8h;
typedef __attribute__((ext_vector_type(8)))  float    v8f;
typedef __attribute__((ext_vector_type(4)))  unsigned int u32x4;
typedef __attribute__((ext_vector_type(8)))  int      i32x8;
typedef __attribute__((ext_vector_type(4)))  int      i32x4;

#define DIMX   1024
#define HEADS  16
#define DHEAD  64
#define FFI    4096
#define SEQ    2048
#define BATCH  2
#define ROWS   (BATCH*SEQ)   // 4096

static __device__ __forceinline__ v16h mk16(v8h lo, v8h hi) {
  return __builtin_shufflevector(lo, hi, 0,1,2,3,4,5,6,7,8,9,10,11,12,13,14,15);
}
static __device__ __forceinline__ v8f wmma16(v16h a, v16h b, v8f c) {
  // D = A(16x32 f16) * B(32x16 f16) + C(16x16 f32)
  return __builtin_amdgcn_wmma_f32_16x16x32_f16(false, a, false, b, (short)0, c, false, false);
}

// ---- Tensor Data Mover: 2D f16 tile load (global -> LDS), wave-level DMA.
// D# group0: [1:0]=count, [63:32]=lds_addr, [120:64]=global_addr, [127:126]=type(2)
// D# group1: [17:16]=data_size(1=2B), [79:48]=tensor_dim0, [111:80]=tensor_dim1,
//            [127:112]=tile_dim0, [143:128]=tile_dim1, [207:160]=tensor_dim0_stride
static __device__ __forceinline__ void tdm_load_tile_f16(
    void* lds_dst, const _Float16* gsrc,
    unsigned tensor_d0, unsigned tensor_d1, unsigned tile_d0, unsigned tile_d1,
    unsigned stride0)
{
  unsigned long long ga = (unsigned long long)(size_t)gsrc;
  unsigned lds_off = (unsigned)(size_t)lds_dst;
  u32x4 g0;
  g0[0] = 1u;                                             // count=1, no gather
  g0[1] = lds_off;                                        // lds_addr
  g0[2] = (unsigned)(ga & 0xffffffffu);                   // global_addr[31:0]
  g0[3] = (unsigned)((ga >> 32) & 0x01ffffffu) | 0x80000000u; // addr[56:32] | type=2
  i32x8 g1;
  g1[0] = 0x00010000;                                     // data_size = 2 bytes
  g1[1] = (int)((tensor_d0 & 0xffffu) << 16);             // tensor_dim0[15:0]
  g1[2] = (int)((tensor_d0 >> 16) | ((tensor_d1 & 0xffffu) << 16));
  g1[3] = (int)((tensor_d1 >> 16) | (tile_d0 << 16));     // tensor_dim1[31:16] | tile_dim0
  g1[4] = (int)(tile_d1 & 0xffffu);                       // tile_dim1 (tile_dim2=0)
  g1[5] = (int)stride0;                                   // tensor_dim0_stride[31:0]
  g1[6] = 0;
  g1[7] = 0;
  i32x4 gz = {0, 0, 0, 0};
  asm volatile("tensor_load_to_lds %0, %1, %2, %3"
               :: "s"(g0), "s"(g1), "s"(gz), "s"(gz)
               : "memory");
}

// ---------------------------------------------------------------- layernorm
__global__ void layernorm_k(const float* __restrict__ x,
                            const float* __restrict__ gamma,
                            _Float16* __restrict__ xn)
{
  __shared__ float red[256];
  const int row = blockIdx.x;
  const int tid = threadIdx.x;
  const float* xr = x + (size_t)row * DIMX;
  float s = 0.f, s2 = 0.f;
  float vals[4];
#pragma unroll
  for (int i = 0; i < 4; ++i) {
    float v = xr[tid + 256*i];
    vals[i] = v; s += v; s2 += v*v;
  }
  red[tid] = s; __syncthreads();
  for (int o = 128; o > 0; o >>= 1) { if (tid < o) red[tid] += red[tid+o]; __syncthreads(); }
  float mu = red[0] * (1.0f / DIMX);
  __syncthreads();
  red[tid] = s2; __syncthreads();
  for (int o = 128; o > 0; o >>= 1) { if (tid < o) red[tid] += red[tid+o]; __syncthreads(); }
  float var = red[0] * (1.0f / DIMX) - mu * mu;
  float rs = rsqrtf(var + 1e-5f);
#pragma unroll
  for (int i = 0; i < 4; ++i) {
    int c = tid + 256*i;
    xn[(size_t)row * DIMX + c] = (_Float16)((vals[i] - mu) * rs * gamma[c]);
  }
}

// ---------------------------------------------------------------- f32 -> f16
__global__ void f32_to_f16_k(const float* __restrict__ in, _Float16* __restrict__ out, size_t n)
{
  size_t i = (size_t)blockIdx.x * blockDim.x + threadIdx.x;
  size_t stride = (size_t)gridDim.x * blockDim.x;
  for (; i < n; i += stride) out[i] = (_Float16)in[i];
}

// ---------------------------------------------------------------- GEMM (f16 in, WMMA, f32 acc)
// C[M,N] = A[M,K] * B[K,N]; block tile 128x64, BK=64, 8 waves, each wave 32x32.
// A tile staged by the Tensor Data Mover; B tile staged transposed by VALU threads.
template<int OUT_HALF>
__global__ void gemm_f16(const _Float16* __restrict__ A, const _Float16* __restrict__ B,
                         float* __restrict__ Cf, _Float16* __restrict__ Ch,
                         int M, int N, int K, float scale)
{
  __shared__ __align__(16) _Float16 lds_a[128 * 64];  // [row][k] 16KB
  __shared__ __align__(16) _Float16 lds_bt[64 * 64];  // [col][k] 8KB (transposed)
  const int tid  = threadIdx.x;
  const int lane = tid & 31;
  const int wid  = tid >> 5;
  const int lg   = lane >> 4;   // lane group 0/1
  const int ll   = lane & 15;
  const int m0   = blockIdx.y * 128;
  const int n0   = blockIdx.x * 64;
  const int wm0  = (wid >> 1) * 32;  // 4 waves along M
  const int wn0  = (wid & 1) * 32;   // 2 waves along N

  v8f acc[2][2] = {};

  for (int k0 = 0; k0 < K; k0 += 64) {
    __syncthreads();
    if (wid == 0) {
      // A tile 128 rows x 64 halfs via TDM (tile_dim0 along contiguous K)
      tdm_load_tile_f16(lds_a, A + (size_t)m0 * K + k0,
                        (unsigned)K, (unsigned)M, 64u, 128u, (unsigned)K);
    }
    { // stage B tile transposed: 64 k-rows x 64 cols
      int row   = tid >> 2;
      int chunk = (tid & 3) * 16;
      const _Float16* s = B + (size_t)(k0 + row) * N + n0 + chunk;
      v8h b0 = *(const v8h*)(s);
      v8h b1 = *(const v8h*)(s + 8);
#pragma unroll
      for (int e = 0; e < 8; ++e) lds_bt[(chunk + e) * 64 + row]     = b0[e];
#pragma unroll
      for (int e = 0; e < 8; ++e) lds_bt[(chunk + 8 + e) * 64 + row] = b1[e];
      if (k0 + 64 < K) // prefetch next B tile slice -> global_prefetch_b8
        __builtin_prefetch(B + (size_t)(k0 + 64 + row) * N + n0 + chunk, 0, 1);
    }
    if (wid == 0) __builtin_amdgcn_s_wait_tensorcnt(0);
    __syncthreads();

#pragma unroll
    for (int s = 0; s < 2; ++s) { // two 32-wide K substeps
      v16h af[2], bf[2];
#pragma unroll
      for (int i = 0; i < 2; ++i) { // A fragment: row=lane%16, k-groups per ISA layout
        int row = wm0 + 16*i + ll;
        af[i] = mk16(*(const v8h*)&lds_a[row*64 + s*32 + 8*lg],
                     *(const v8h*)&lds_a[row*64 + s*32 + 16 + 8*lg]);
      }
#pragma unroll
      for (int j = 0; j < 2; ++j) { // B fragment: col=lane%16, k-groups per ISA layout
        int col = wn0 + 16*j + ll;
        bf[j] = mk16(*(const v8h*)&lds_bt[col*64 + s*32 + 8*lg],
                     *(const v8h*)&lds_bt[col*64 + s*32 + 16 + 8*lg]);
      }
#pragma unroll
      for (int i = 0; i < 2; ++i)
#pragma unroll
        for (int j = 0; j < 2; ++j)
          acc[i][j] = wmma16(af[i], bf[j], acc[i][j]);
    }
  }

  // epilogue: C layout -> VGPR v holds row (v + 8*lg), col = lane%16
#pragma unroll
  for (int i = 0; i < 2; ++i)
#pragma unroll
    for (int j = 0; j < 2; ++j)
#pragma unroll
      for (int v = 0; v < 8; ++v) {
        int row = m0 + wm0 + 16*i + v + 8*lg;
        int col = n0 + wn0 + 16*j + ll;
        float xv = acc[i][j][v] * scale;
        if (OUT_HALF) Ch[(size_t)row * N + col] = (_Float16)xv;
        else          Cf[(size_t)row * N + col] = xv;
      }
}

// ---------------------------------------------------------------- flash attention (multi-query)
// q: [ROWS, HEADS*DHEAD] f16 (pre-scaled); kv: [ROWS, 128] f16 (K in [0,64), V in [64,128))
// o: [ROWS, HEADS*DHEAD] f16
__global__ void flash_attn_k(const _Float16* __restrict__ qh,
                             const _Float16* __restrict__ kvh,
                             _Float16* __restrict__ oh)
{
  __shared__ __align__(16) _Float16 smem[4 * (16*32 + 64*32)];
  const int tid  = threadIdx.x;
  const int lane = tid & 31;
  const int wid  = tid >> 5;
  const int lg   = lane >> 4;
  const int ll   = lane & 15;
  const int head = blockIdx.y;
  const int bb   = blockIdx.z;
  const int qblk = blockIdx.x * 4 + wid;

  _Float16* P  = smem + wid * (16*32 + 64*32); // 16x32 probs
  _Float16* Vt = P + 16*32;                    // [d][key] 64x32

  // Q fragments (A-matrix 16x32 layout), K-dim split 0..31 / 32..63
  const _Float16* qrow = qh + ((size_t)bb*SEQ + qblk*16 + ll) * (HEADS*DHEAD) + head*DHEAD;
  v16h aq[2];
#pragma unroll
  for (int kk = 0; kk < 2; ++kk)
    aq[kk] = mk16(*(const v8h*)(qrow + kk*32 + 8*lg),
                  *(const v8h*)(qrow + kk*32 + 16 + 8*lg));

  v8f o[4] = {};
  float m[8], l[8];
#pragma unroll
  for (int v = 0; v < 8; ++v) { m[v] = -1e30f; l[v] = 0.f; }

  const _Float16* kvb = kvh + (size_t)bb * SEQ * 128;

  for (int kb = 0; kb < SEQ/32; ++kb) {
    __syncthreads();
    { // stage V^T for this 32-key block: lane <-> key
      const _Float16* vr = kvb + (size_t)(kb*32 + lane) * 128 + 64;
#pragma unroll
      for (int c = 0; c < 64; c += 8) {
        v8h vv = *(const v8h*)(vr + c);
#pragma unroll
        for (int e = 0; e < 8; ++e) Vt[(c+e)*32 + lane] = vv[e];
      }
    }
    // sim 16x32 = Q(16x64) K^T(64x32): two N-tiles, chained over K=64
    v8f s0 = {}, s1 = {};
#pragma unroll
    for (int kk = 0; kk < 2; ++kk) {
      const _Float16* kr0 = kvb + (size_t)(kb*32 + ll)      * 128 + kk*32 + 8*lg;
      const _Float16* kr1 = kvb + (size_t)(kb*32 + 16 + ll) * 128 + kk*32 + 8*lg;
      v16h b0 = mk16(*(const v8h*)(kr0), *(const v8h*)(kr0 + 16));
      v16h b1 = mk16(*(const v8h*)(kr1), *(const v8h*)(kr1 + 16));
      s0 = wmma16(aq[kk], b0, s0);
      s1 = wmma16(aq[kk], b1, s1);
    }
    // online softmax; VGPR slot v holds row (v + 8*lg), 16 lanes per row
#pragma unroll
    for (int v = 0; v < 8; ++v) {
      float mx = fmaxf(s0[v], s1[v]);
#pragma unroll
      for (int msk = 8; msk >= 1; msk >>= 1) mx = fmaxf(mx, __shfl_xor(mx, msk, 32));
      float mn = fmaxf(m[v], mx);
      float alpha = __expf(m[v] - mn);
      float p0 = __expf(s0[v] - mn);
      float p1 = __expf(s1[v] - mn);
      float ps = p0 + p1;
#pragma unroll
      for (int msk = 8; msk >= 1; msk >>= 1) ps += __shfl_xor(ps, msk, 32);
      l[v] = l[v] * alpha + ps;
      m[v] = mn;
#pragma unroll
      for (int cg = 0; cg < 4; ++cg) o[cg][v] *= alpha;
      int prow = v + 8*lg;
      P[prow*32 + ll]      = (_Float16)p0;
      P[prow*32 + 16 + ll] = (_Float16)p1;
    }
    __syncthreads();
    // O(16x64) += P(16x32) V(32x64)
    v16h pf = mk16(*(const v8h*)&P[ll*32 + 8*lg],
                   *(const v8h*)&P[ll*32 + 16 + 8*lg]);
#pragma unroll
    for (int cg = 0; cg < 4; ++cg) {
      int n = cg*16 + ll;
      v16h vf = mk16(*(const v8h*)&Vt[n*32 + 8*lg],
                     *(const v8h*)&Vt[n*32 + 16 + 8*lg]);
      o[cg] = wmma16(pf, vf, o[cg]);
    }
  }

#pragma unroll
  for (int v = 0; v < 8; ++v) {
    float rl = 1.f / l[v];
    int row = qblk*16 + v + 8*lg;
    _Float16* orow = oh + ((size_t)bb*SEQ + row) * (HEADS*DHEAD) + head*DHEAD;
#pragma unroll
    for (int cg = 0; cg < 4; ++cg)
      orow[cg*16 + ll] = (_Float16)(o[cg][v] * rl);
  }
}

// ---------------------------------------------------------------- SwiGLU
__global__ void swiglu_k(const _Float16* __restrict__ h, _Float16* __restrict__ g)
{
  size_t idx = (size_t)blockIdx.x * blockDim.x + threadIdx.x;
  size_t stride = (size_t)gridDim.x * blockDim.x;
  const size_t total = (size_t)ROWS * FFI;
  for (; idx < total; idx += stride) {
    size_t row = idx / FFI;
    int col = (int)(idx - row * FFI);
    float val = (float)h[row * (2*FFI) + col];
    float gt  = (float)h[row * (2*FFI) + FFI + col];
    float si  = gt / (1.f + __expf(-gt));
    g[idx] = (_Float16)(si * val);
  }
}

// ---------------------------------------------------------------- final add
__global__ void add_k(const float* __restrict__ a, const float* __restrict__ b,
                      float* __restrict__ o, size_t n)
{
  size_t i = (size_t)blockIdx.x * blockDim.x + threadIdx.x;
  size_t stride = (size_t)gridDim.x * blockDim.x;
  for (; i < n; i += stride) o[i] = a[i] + b[i];
}

// ================================================================ launch
extern "C" void kernel_launch(void* const* d_in, const int* in_sizes, int n_in,
                              void* d_out, int out_size, void* d_ws, size_t ws_size,
                              hipStream_t stream) {
  const float* x     = (const float*)d_in[0];
  const float* gamma = (const float*)d_in[1];
  const float* Wq    = (const float*)d_in[2];
  const float* Wkv   = (const float*)d_in[3];
  const float* Wo    = (const float*)d_in[4];
  const float* Wff1  = (const float*)d_in[5];
  const float* Wff2  = (const float*)d_in[6];
  float* out = (float*)d_out;

  char* p = (char*)d_ws;
  auto alloc = [&](size_t bytes) -> char* {
    char* r = p;
    p += (bytes + 255) & ~(size_t)255;
    return r;
  };
  _Float16* Wq_h   = (_Float16*)alloc((size_t)DIMX * (HEADS*DHEAD) * 2);
  _Float16* Wkv_h  = (_Float16*)alloc((size_t)DIMX * 128 * 2);
  _Float16* Wo_h   = (_Float16*)alloc((size_t)(HEADS*DHEAD) * DIMX * 2);
  _Float16* Wff1_h = (_Float16*)alloc((size_t)DIMX * (2*FFI) * 2);
  _Float16* Wff2_h = (_Float16*)alloc((size_t)FFI * DIMX * 2);
  _Float16* xn_h   = (_Float16*)alloc((size_t)ROWS * DIMX * 2);
  _Float16* q_h    = (_Float16*)alloc((size_t)ROWS * (HEADS*DHEAD) * 2);
  _Float16* kv_h   = (_Float16*)alloc((size_t)ROWS * 128 * 2);
  _Float16* attn_h = (_Float16*)alloc((size_t)ROWS * (HEADS*DHEAD) * 2);
  _Float16* h_h    = (_Float16*)alloc((size_t)ROWS * (2*FFI) * 2);
  _Float16* g_h    = (_Float16*)alloc((size_t)ROWS * FFI * 2);
  float*    oattn  = (float*)   alloc((size_t)ROWS * DIMX * 4);
  float*    off    = (float*)   alloc((size_t)ROWS * DIMX * 4);

  // 1) weight conversions fp32 -> f16 (one-time per launch; traffic-trivial)
  f32_to_f16_k<<<2048, 256, 0, stream>>>(Wq,   Wq_h,   (size_t)DIMX * (HEADS*DHEAD));
  f32_to_f16_k<<<512,  256, 0, stream>>>(Wkv,  Wkv_h,  (size_t)DIMX * 128);
  f32_to_f16_k<<<2048, 256, 0, stream>>>(Wo,   Wo_h,   (size_t)(HEADS*DHEAD) * DIMX);
  f32_to_f16_k<<<4096, 256, 0, stream>>>(Wff1, Wff1_h, (size_t)DIMX * (2*FFI));
  f32_to_f16_k<<<4096, 256, 0, stream>>>(Wff2, Wff2_h, (size_t)FFI * DIMX);

  // 2) LayerNorm -> xn (f16)
  layernorm_k<<<ROWS, 256, 0, stream>>>(x, gamma, xn_h);

  // 3) projections (WMMA GEMMs, TDM-staged A tiles); Q pre-scaled by 1/sqrt(DHEAD)
  gemm_f16<1><<<dim3((HEADS*DHEAD)/64, ROWS/128), 256, 0, stream>>>(
      xn_h, Wq_h, nullptr, q_h, ROWS, HEADS*DHEAD, DIMX, 0.125f);
  gemm_f16<1><<<dim3(128/64, ROWS/128), 256, 0, stream>>>(
      xn_h, Wkv_h, nullptr, kv_h, ROWS, 128, DIMX, 1.0f);
  gemm_f16<1><<<dim3((2*FFI)/64, ROWS/128), 256, 0, stream>>>(
      xn_h, Wff1_h, nullptr, h_h, ROWS, 2*FFI, DIMX, 1.0f);

  // 4) flash attention (multi-query: one K/V head)
  flash_attn_k<<<dim3(SEQ/16/4, HEADS, BATCH), 128, 0, stream>>>(q_h, kv_h, attn_h);

  // 5) output projection -> fp32
  gemm_f16<0><<<dim3(DIMX/64, ROWS/128), 256, 0, stream>>>(
      attn_h, Wo_h, oattn, nullptr, ROWS, DIMX, HEADS*DHEAD, 1.0f);

  // 6) SwiGLU -> g (f16), then FF2 -> fp32
  swiglu_k<<<4096, 256, 0, stream>>>(h_h, g_h);
  gemm_f16<0><<<dim3(DIMX/64, ROWS/128), 256, 0, stream>>>(
      g_h, Wff2_h, off, nullptr, ROWS, DIMX, FFI, 1.0f);

  // 7) out = attn_out + ff_out
  add_k<<<4096, 256, 0, stream>>>(oattn, off, out, (size_t)ROWS * DIMX);
}
